// RWKV_Time_Mixing_v4_60679297958536
// MI455X (gfx1250) — compile-verified, matrix operand order checked by reading, not run on hardware
//
#include <hip/hip_runtime.h>
#include <hip/hip_bf16.h>

typedef __attribute__((ext_vector_type(16))) __bf16    v16bf;
typedef __attribute__((ext_vector_type(8)))  __bf16    v8bf;
typedef __attribute__((ext_vector_type(8)))  float     v8f;
typedef __attribute__((ext_vector_type(4)))  unsigned  v4u;
typedef __attribute__((ext_vector_type(8)))  int       v8i;
typedef __attribute__((ext_vector_type(4)))  int       v4i;

constexpr int kB = 4;
constexpr int kT = 2048;
constexpr int kC = 2048;
constexpr int kH = 2048;
constexpr int kM = kB * kT;   // 8192 rows of activations

// GEMM tiling
constexpr int MBLK = 128;     // block tile M
constexpr int NBLK = 128;     // block tile N
constexpr int KBLK = 64;      // K slab staged in LDS per stage
constexpr int LDT  = 72;      // padded LDS row stride (elements) -> 144B, 16B aligned

#define HAVE_TDM __has_builtin(__builtin_amdgcn_tensor_load_to_lds)

// ---------------------------------------------------------------------------
// WMMA fragment load from an LDS tile (row-major, leading dim LDT).
// CDNA5 16-bit A/B layout: lane L -> row (L&15), half = L>>4; the lane's 16
// elements are K = half*8 + [0..7] and K + 16 + half*8 + [0..7]  => two
// contiguous 16B LDS reads per lane.
// ---------------------------------------------------------------------------
__device__ __forceinline__ v16bf lds_frag(const __bf16* tile, int row_base, int k_base) {
  const int lane = threadIdx.x & 31;
  const int row  = row_base + (lane & 15);
  const int half = lane >> 4;
  const __bf16* base = tile + row * LDT + k_base + half * 8;
  v8bf lo = *(const v8bf*)(base);
  v8bf hi = *(const v8bf*)(base + 16);
  return __builtin_shufflevector(lo, hi, 0, 1, 2, 3, 4, 5, 6, 7,
                                         8, 9, 10, 11, 12, 13, 14, 15);
}

// C/D 16x16 f32 layout: lane L -> col (L&15), VGPR r -> row (L>>4)*8 + r.
__device__ __forceinline__ void store_tile16x16(float* __restrict__ D, int ldd,
                                                int m_base, int n_base, v8f c) {
  const int lane = threadIdx.x & 31;
  const int n = n_base + (lane & 15);
  const int m = m_base + (lane >> 4) * 8;
#pragma unroll
  for (int r = 0; r < 8; ++r) D[(size_t)(m + r) * ldd + n] = c[r];
}

#if HAVE_TDM
// ---------------------------------------------------------------------------
// Tensor Data Mover staging: one TENSOR_LOAD_TO_LDS moves a whole
// (tile_rows x KBLK) bf16 slab. D# per CDNA5 ISA:
//   group0: count=1 | lds_addr | global_addr(57b) | type=2
//   group1: data_size=1(2B), pad_enable, pad_interval=4 (32 DWORDs = one
//           128B row), pad_amount=3 (4 DWORDs = 16B) -> hardware produces the
//           144B padded rows (LDT=72) the compute side expects.
//   tensor_dim0=K, dim0_stride=K (elements), tile_dim0=KBLK, tile_dim1=rows.
// Issued by one wave; tracked by TENSORcnt. Groups 2/3 (dims 2..4) unused
// for a 2D tile -> zero-filled.
// ---------------------------------------------------------------------------
__device__ __forceinline__ void tdm_load_tile(const __bf16* __restrict__ gsrc,
                                              unsigned lds_off, int rows_total,
                                              int K, int row_base, int k0,
                                              int tile_rows) {
  const unsigned long long gaddr =
      (unsigned long long)(size_t)(gsrc + (size_t)row_base * K + k0);
  v4u g0;
  g0.x = 1u;                                        // count=1, user descriptor
  g0.y = lds_off;                                   // lds_addr
  g0.z = (unsigned)(gaddr & 0xffffffffu);           // global_addr[31:0]
  g0.w = (unsigned)(gaddr >> 32) | (2u << 30);      // global_addr[56:32] | type=2

  v8i g1;
  g1[0] = (1 << 16)            // data_size = 2 bytes
        | (1 << 20)            // pad_enable
        | (4 << 22)            // pad_interval: 2^(4+1)=32 DWORDs (128B row)
        | (3 << 25);           // pad_amount: 4 DWORDs (16B)
  g1[1] = (K & 0xffff) << 16;                                   // tensor_dim0 lo
  g1[2] = (int)(((unsigned)K >> 16) |
                (((unsigned)rows_total & 0xffffu) << 16));      // dim0 hi | dim1 lo
  g1[3] = (int)(((unsigned)rows_total >> 16) |
                ((unsigned)KBLK << 16));                        // dim1 hi | tile_dim0
  g1[4] = tile_rows;                                            // tile_dim1 (tile_dim2=0)
  g1[5] = K;                                                    // dim0_stride lo
  g1[6] = 0;                                                    // stride hi / dim1_stride
  g1[7] = 0;
  const v4i z4 = {0, 0, 0, 0};
  const v8i z8 = {0, 0, 0, 0, 0, 0, 0, 0};
  // clang-23 / therock-10.0 arity: (v4u g0, v8i g1, v4i, v4i, v8i, i32 cpol)
  __builtin_amdgcn_tensor_load_to_lds(g0, g1, z4, z4, z8, 0);
}
#else
// Fallback: per-lane async global->LDS copies (ASYNCcnt), 4 chunks/thread/tile.
__device__ __forceinline__ void async_copy_tile(const __bf16* __restrict__ gsrc,
                                                int K, int row_base, int k0,
                                                __bf16* dst) {
  const int tid = threadIdx.x;
#pragma unroll
  for (int i = 0; i < 4; ++i) {
    const int chunk = tid + i * 256;
    const int row = chunk >> 3;
    const int c16 = chunk & 7;
    const __bf16* g = gsrc + (size_t)(row_base + row) * K + k0 + c16 * 8;
    const unsigned l = (unsigned)(size_t)(dst + row * LDT + c16 * 8);
    asm volatile("global_load_async_to_lds_b128 %0, %1, off"
                 :: "v"(l), "v"(g) : "memory");
  }
}
#endif

// Compute both 32-wide K steps of one staged slab. Wave tile = 64(M) x 32(N):
// 4 M-frags x 2 N-frags = 8 WMMAs per K step vs 6 LDS fragment loads.
__device__ __forceinline__ void compute_stage(const __bf16* aT, const __bf16* bT,
                                              int wm, int wn, v8f (&acc)[4][2]) {
#pragma unroll
  for (int ks = 0; ks < KBLK; ks += 32) {
    v16bf a[4], b[2];
#pragma unroll
    for (int mt = 0; mt < 4; ++mt) a[mt] = lds_frag(aT, wm * 64 + mt * 16, ks);
#pragma unroll
    for (int nt = 0; nt < 2; ++nt) b[nt] = lds_frag(bT, wn * 32 + nt * 16, ks);
#pragma unroll
    for (int mt = 0; mt < 4; ++mt)
#pragma unroll
      for (int nt = 0; nt < 2; ++nt)
        acc[mt][nt] = __builtin_amdgcn_wmma_f32_16x16x32_bf16(
            false, a[mt], false, b[nt], (short)0, acc[mt][nt], false, false);
  }
}

// ---------------------------------------------------------------------------
// GEMM: D[M,N] = A[M,K] * W[N,K]^T  (A, W bf16 row-major; D f32)
// 256 threads = 8 waves (2 M x 4 N), block tile 128x128, double-buffered
// TDM (or async-to-LDS) staging over K in slabs of 64.
// ---------------------------------------------------------------------------
__global__ __launch_bounds__(256, 2) void gemm_bf16_lds(
    const __bf16* __restrict__ A, const __bf16* __restrict__ W,
    float* __restrict__ D, int M, int N, int K) {
  __shared__ __align__(16) __bf16 sA[2][MBLK * LDT];
  __shared__ __align__(16) __bf16 sB[2][NBLK * LDT];

  const int wave = threadIdx.x >> 5;
  const int wm = wave >> 2;      // 0..1  (64 rows each)
  const int wn = wave & 3;       // 0..3  (32 cols each)
  const int mB = blockIdx.x * MBLK;
  const int nB = blockIdx.y * NBLK;

  v8f acc[4][2] = {};
  const int nstage = K / KBLK;

#if HAVE_TDM
  const bool isA = (wave == 0);        // wave 0 DMAs the A tile
  const bool isB = (wave == 1);        // wave 1 DMAs the B tile
  if (isA) tdm_load_tile(A, (unsigned)(size_t)&sA[0][0], M, K, mB, 0, MBLK);
  if (isB) tdm_load_tile(W, (unsigned)(size_t)&sB[0][0], N, K, nB, 0, NBLK);

  for (int s = 0; s < nstage; ++s) {
    if (s + 1 < nstage) {
      const int nb = (s + 1) & 1;
      const int k1 = (s + 1) * KBLK;
      if (isA) tdm_load_tile(A, (unsigned)(size_t)&sA[nb][0], M, K, mB, k1, MBLK);
      if (isB) tdm_load_tile(W, (unsigned)(size_t)&sB[nb][0], N, K, nB, k1, NBLK);
      // 1 outstanding TDM (stage s+1) per issuing wave => stage s landed.
      __builtin_amdgcn_s_wait_tensorcnt(1);
    } else {
      __builtin_amdgcn_s_wait_tensorcnt(0);
    }
    __syncthreads();                       // tile visible to all waves
    compute_stage(sA[s & 1], sB[s & 1], wm, wn, acc);
    __syncthreads();                       // done reading before next overwrite
  }
#else
  async_copy_tile(A, K, mB, 0, sA[0]);
  async_copy_tile(W, K, nB, 0, sB[0]);
  for (int s = 0; s < nstage; ++s) {
    if (s + 1 < nstage) {
      const int nb = (s + 1) & 1;
      async_copy_tile(A, K, mB, (s + 1) * KBLK, sA[nb]);
      async_copy_tile(W, K, nB, (s + 1) * KBLK, sB[nb]);
      asm volatile("s_wait_asynccnt 0x8" ::: "memory");
    } else {
      asm volatile("s_wait_asynccnt 0x0" ::: "memory");
    }
    __syncthreads();
    compute_stage(sA[s & 1], sB[s & 1], wm, wn, acc);
    __syncthreads();
  }
#endif

  const int m0 = mB + wm * 64;
  const int n0 = nB + wn * 32;
#pragma unroll
  for (int mt = 0; mt < 4; ++mt)
#pragma unroll
    for (int nt = 0; nt < 2; ++nt)
      store_tile16x16(D, N, m0 + mt * 16, n0 + nt * 16, acc[mt][nt]);
}

// ---------------------------------------------------------------------------
// Time-shift + token mixing, fp32 -> bf16.  shifted[t] = x[t-1], 0 at t=0.
// ---------------------------------------------------------------------------
__global__ __launch_bounds__(256) void prep_mix_kernel(
    const float* __restrict__ x,
    const float* __restrict__ mu_r, const float* __restrict__ mu_k,
    const float* __restrict__ mu_v,
    __bf16* __restrict__ xr, __bf16* __restrict__ xk, __bf16* __restrict__ xv) {
  size_t idx = (size_t)blockIdx.x * blockDim.x + threadIdx.x;
  const size_t total = (size_t)kB * kT * kC;
  if (idx >= total) return;
  const int c = (int)(idx % kC);
  const int t = (int)((idx / kC) % kT);
  const float xc = x[idx];
  const float sh = (t > 0) ? x[idx - kC] : 0.0f;
  const float mr = mu_r[c], mk = mu_k[c], mv = mu_v[c];
  xr[idx] = (__bf16)(mr * xc + (1.0f - mr) * sh);
  xk[idx] = (__bf16)(mk * xc + (1.0f - mk) * sh);
  xv[idx] = (__bf16)(mv * xc + (1.0f - mv) * sh);
}

// fp32 -> bf16 weight conversion for all four weight matrices.
__global__ __launch_bounds__(256) void convert_w_kernel(
    const float* __restrict__ Wr, const float* __restrict__ Wk,
    const float* __restrict__ Wv, const float* __restrict__ Wo,
    __bf16* __restrict__ br, __bf16* __restrict__ bk,
    __bf16* __restrict__ bv, __bf16* __restrict__ bo) {
  size_t i = (size_t)blockIdx.x * blockDim.x + threadIdx.x;
  const size_t total = (size_t)kH * kC;
  if (i >= total) return;
  br[i] = (__bf16)Wr[i];
  bk[i] = (__bf16)Wk[i];
  bv[i] = (__bf16)Wv[i];
  bo[i] = (__bf16)Wo[i];
}

// ---------------------------------------------------------------------------
// RWKV-v4 stabilized WKV recurrence fused with sigmoid(r) * y -> bf16 act.
// One thread per (b, h) channel; warp accesses 32 consecutive h -> coalesced.
// ---------------------------------------------------------------------------
__global__ __launch_bounds__(256) void wkv_act_kernel(
    const float* __restrict__ r, const float* __restrict__ k,
    const float* __restrict__ v, const float* __restrict__ w,
    const float* __restrict__ u, __bf16* __restrict__ act) {
  const int idx = blockIdx.x * blockDim.x + threadIdx.x;  // 0 .. B*H-1
  const int h = idx % kH;
  const int b = idx / kH;
  const float weff = -expf(w[h]);
  const float uu = u[h];

  float aa = 0.0f, bb = 0.0f, pp = -1e38f;
  const size_t base = (size_t)b * kT * kH + h;
  for (int t = 0; t < kT; ++t) {
    const size_t i = base + (size_t)t * kH;
    const float kt = k[i];
    const float vt = v[i];
    const float rt = r[i];

    const float ww = uu + kt;
    const float p  = fmaxf(pp, ww);
    const float e1 = expf(pp - p);
    const float e2 = expf(ww - p);
    const float y  = (e1 * aa + e2 * vt) / (e1 * bb + e2);

    const float ww2 = weff + pp;
    const float p2  = fmaxf(ww2, kt);
    const float e1b = expf(ww2 - p2);
    const float e2b = expf(kt - p2);
    aa = e1b * aa + e2b * vt;
    bb = e1b * bb + e2b;
    pp = p2;

    const float s = 1.0f / (1.0f + expf(-rt));
    act[i] = (__bf16)(s * y);
  }
}

// ---------------------------------------------------------------------------
// Host-side launcher.
// ---------------------------------------------------------------------------
extern "C" void kernel_launch(void* const* d_in, const int* in_sizes, int n_in,
                              void* d_out, int out_size, void* d_ws, size_t ws_size,
                              hipStream_t stream) {
  (void)in_sizes; (void)n_in; (void)out_size; (void)ws_size;

  const float* x     = (const float*)d_in[0];
  const float* mu_r  = (const float*)d_in[1];
  const float* mu_k  = (const float*)d_in[2];
  const float* mu_v  = (const float*)d_in[3];
  const float* wkv_w = (const float*)d_in[4];
  const float* wkv_u = (const float*)d_in[5];
  const float* Wr    = (const float*)d_in[6];
  const float* Wk    = (const float*)d_in[7];
  const float* Wv    = (const float*)d_in[8];
  const float* Wo    = (const float*)d_in[9];
  float* out = (float*)d_out;

  const size_t WN = (size_t)kH * kC;   // weight elements
  const size_t MK = (size_t)kM * kC;   // activation elements

  char* ws = (char*)d_ws;
  __bf16* Wr_b = (__bf16*)ws;              ws += WN * 2;
  __bf16* Wk_b = (__bf16*)ws;              ws += WN * 2;
  __bf16* Wv_b = (__bf16*)ws;              ws += WN * 2;
  __bf16* Wo_b = (__bf16*)ws;              ws += WN * 2;
  __bf16* xr_b = (__bf16*)ws;              ws += MK * 2;
  __bf16* xk_b = (__bf16*)ws;              ws += MK * 2;
  __bf16* xv_b = (__bf16*)ws;              ws += MK * 2;
  float*  r_f  = (float*)ws;               ws += MK * 4;
  float*  k_f  = (float*)ws;               ws += MK * 4;
  float*  v_f  = (float*)ws;               ws += MK * 4;
  __bf16* act_b = xr_b;  // xr consumed after first GEMM; reuse for activations

  // 1) weight conversion + mixing
  {
    const int thr = 256;
    convert_w_kernel<<<(int)((WN + thr - 1) / thr), thr, 0, stream>>>(
        Wr, Wk, Wv, Wo, Wr_b, Wk_b, Wv_b, Wo_b);
    prep_mix_kernel<<<(int)((MK + thr - 1) / thr), thr, 0, stream>>>(
        x, mu_r, mu_k, mu_v, xr_b, xk_b, xv_b);
  }

  // 2) r / k / v projections
  {
    dim3 grid(kM / MBLK, kH / NBLK);
    dim3 block(256);
    gemm_bf16_lds<<<grid, block, 0, stream>>>(xr_b, Wr_b, r_f, kM, kH, kC);
    gemm_bf16_lds<<<grid, block, 0, stream>>>(xk_b, Wk_b, k_f, kM, kH, kC);
    gemm_bf16_lds<<<grid, block, 0, stream>>>(xv_b, Wv_b, v_f, kM, kH, kC);
  }

  // 3) WKV recurrence fused with sigmoid(r) gating -> bf16 activations
  wkv_act_kernel<<<(kB * kH) / 256, 256, 0, stream>>>(
      r_f, k_f, v_f, wkv_w, wkv_u, act_b);

  // 4) output projection
  {
    dim3 grid(kM / MBLK, kC / NBLK);
    dim3 block(256);
    gemm_bf16_lds<<<grid, block, 0, stream>>>(act_b, Wo_b, out, kM, kC, kH);
  }
}
